// Region_Aware_Spatial_Attention_85031762526229
// MI455X (gfx1250) — matrix operand order repaired and checked by reading, not compiled, not run on hardware
//
#include <hip/hip_runtime.h>
#include <hip/hip_bf16.h>

// ---------------------------------------------------------------------------
// Region-Aware Spatial Attention for MI455X (gfx1250, wave32, WMMA bf16)
// B=4 H=4 N=4096 D=C=256, FFN 256->128->64->1
// ---------------------------------------------------------------------------

#define BB 4
#define HH 4
#define NN 4096
#define DD 256
#define BN (BB * NN)          // 16384 tokens
#define KO 1024               // H*D for output projection
#define EPS 1e-5f
#define LOG2E 1.44269504088896f

typedef __bf16 bf16_t;
typedef __attribute__((ext_vector_type(16))) bf16_t v16bf;
typedef __attribute__((ext_vector_type(8)))  float  v8f;
typedef __attribute__((ext_vector_type(4)))  unsigned int v4u;

union Frag { v16bf v; v4u u[2]; };

__device__ inline bf16_t f2bf(float f) {
    unsigned u = __builtin_bit_cast(unsigned, f);
    u += 0x7FFFu + ((u >> 16) & 1u);            // round-to-nearest-even
    unsigned short s = (unsigned short)(u >> 16);
    return __builtin_bit_cast(bf16_t, s);
}

__device__ inline v8f wmma_bf16(const v16bf& a, const v16bf& b, const v8f& c) {
    return __builtin_amdgcn_wmma_f32_16x16x32_bf16(false, a, false, b, (short)0, c,
                                                   false, false);
}

// ---------------------------------------------------------------------------
// Elementwise f32 -> bf16 convert
// ---------------------------------------------------------------------------
__global__ void cvt_bf16(const float* __restrict__ src, bf16_t* __restrict__ dst,
                         size_t n) {
    for (size_t i = (size_t)blockIdx.x * blockDim.x + threadIdx.x; i < n;
         i += (size_t)gridDim.x * blockDim.x)
        dst[i] = f2bf(src[i]);
}

// ---------------------------------------------------------------------------
// img_emb [B,C,G,G] -> x_img [B,N,C] (bf16) and x_imgpos = img + pos (bf16)
// ---------------------------------------------------------------------------
__global__ __launch_bounds__(256) void transpose_img(const float* __restrict__ img_emb,
                                                     const float* __restrict__ pos,
                                                     bf16_t* __restrict__ x_img,
                                                     bf16_t* __restrict__ x_imgpos) {
    __shared__ float tile[32][33];
    const int b  = blockIdx.z;
    const int cb = blockIdx.y * 32;
    const int nb = blockIdx.x * 32;
    const int tx = threadIdx.x, ty = threadIdx.y;
#pragma unroll
    for (int i = 0; i < 4; ++i) {
        int c = cb + ty + i * 8;
        tile[ty + i * 8][tx] = img_emb[((size_t)b * DD + c) * NN + nb + tx];
    }
    __syncthreads();
#pragma unroll
    for (int i = 0; i < 4; ++i) {
        int n = nb + ty + i * 8;
        int c = cb + tx;
        float v = tile[tx][ty + i * 8];
        size_t o = ((size_t)b * NN + n) * DD + c;
        x_img[o]    = f2bf(v);
        x_imgpos[o] = f2bf(v + pos[(size_t)n * DD + c]);
    }
}

// ---------------------------------------------------------------------------
// Weight transpose+convert: dst[b][c][r] = bf16(src[b][r][c])
// ---------------------------------------------------------------------------
__global__ void wtrans(const float* __restrict__ src, bf16_t* __restrict__ dst,
                       int rows, int cols, int batch) {
    size_t total = (size_t)batch * rows * cols;
    for (size_t i = (size_t)blockIdx.x * blockDim.x + threadIdx.x; i < total;
         i += (size_t)gridDim.x * blockDim.x) {
        size_t r = i % rows;
        size_t t = i / rows;
        size_t c = t % cols;
        size_t bb = t / cols;
        dst[i] = f2bf(src[(bb * rows + r) * cols + c]);
    }
}

// ---------------------------------------------------------------------------
// Async-copy a 256-row x 512B weight panel into padded LDS (row stride 528B).
// rowStrideBytes: global row stride; colOffBytes: slab offset within each row.
// ---------------------------------------------------------------------------
__device__ __forceinline__ void stage_w_panel(const bf16_t* gbase, int tid,
                                              bf16_t (*sW)[264],
                                              int rowStrideBytes, int colOffBytes) {
    for (int j = 0; j < 64; ++j) {            // 8192 chunks of 16B / 128 threads
        int i = tid + j * 128;
        int row = i >> 5;
        unsigned goff = (unsigned)(row * rowStrideBytes + colOffBytes + (i & 31) * 16);
        unsigned lds  = (unsigned)(size_t)(&sW[row][(i & 31) * 8]);
        asm volatile("global_load_async_to_lds_b128 %0, %1, %2"
                     :: "v"(lds), "v"(goff), "s"(gbase) : "memory");
    }
}

// ---------------------------------------------------------------------------
// Per-head projection GEMM: out[bh] = X[b] (4096x256) @ W[h] (256x256) + bias
// grid = (B*H, N/64), block = 128 (4 waves, 16 rows each).
// Full 128KB weight panel staged to LDS (async); A frags in registers;
// inner loop is pure ds_load_b128 + v_wmma.
// transposed=1 stores out[bh][d][n] (for V), else out[bh][n][d].
// ---------------------------------------------------------------------------
__global__ void __launch_bounds__(128, 1)
proj_gemm(const bf16_t* __restrict__ X, const bf16_t* __restrict__ WT,
          const float* __restrict__ bias, bf16_t* __restrict__ out,
          float scale, int transposed) {
    const int tid = threadIdx.x;
    const int wave = tid >> 5, lane = tid & 31;
    const int half = lane >> 4, ln = lane & 15;
    const int bh = blockIdx.x;
    const int b = bh >> 2, h = bh & 3;
    const int rowbase = blockIdx.y * 64 + wave * 16;

    __shared__ bf16_t sW[256][264];

    const bf16_t* xrow  = X + ((size_t)b * NN + rowbase + ln) * DD;
    const bf16_t* wbase = WT + (size_t)h * DD * DD;

    stage_w_panel(wbase, tid, sW, DD * 2, 0);

    // A fragments (independent of LDS staging)
    Frag af[8];
#pragma unroll
    for (int kc = 0; kc < 8; ++kc) {
        af[kc].u[0] = *(const v4u*)(xrow + kc * 32 + half * 16);
        af[kc].u[1] = *(const v4u*)(xrow + kc * 32 + half * 16 + 8);
    }

    v8f acc[16];
#pragma unroll
    for (int nt = 0; nt < 16; ++nt)
#pragma unroll
        for (int r = 0; r < 8; ++r) acc[nt][r] = 0.f;

    asm volatile("s_wait_asynccnt 0x0" ::: "memory");
    __syncthreads();

#pragma unroll
    for (int kc = 0; kc < 8; ++kc) {
#pragma unroll
        for (int nt = 0; nt < 16; ++nt) {
            const bf16_t* wp = &sW[nt * 16 + ln][kc * 32 + half * 16];
            Frag bf;
            bf.u[0] = *(const v4u*)(wp);
            bf.u[1] = *(const v4u*)(wp + 8);
            acc[nt] = wmma_bf16(af[kc].v, bf.v, acc[nt]);
        }
    }
#pragma unroll
    for (int nt = 0; nt < 16; ++nt) {
        float bv = bias[h * DD + nt * 16 + ln];
#pragma unroll
        for (int r = 0; r < 8; ++r) {
            float val = (acc[nt][r] + bv) * scale;
            int gm = rowbase + 8 * half + r;
            if (transposed)
                out[((size_t)bh * DD + nt * 16 + ln) * NN + gm] = f2bf(val);
            else
                out[((size_t)bh * NN + gm) * DD + nt * 16 + ln] = f2bf(val);
        }
    }
}

// ---------------------------------------------------------------------------
// Flash attention with async-to-LDS K/V staging (double buffered).
// Per workgroup: 4 waves x 16 queries = 64 queries; loop over 32-key tiles.
// sK: 32 rows x 256 bf16 (row stride 264 elem = 528B, conflict-free b128)
// sV: 256 rows x 32 bf16 (row stride  40 elem =  80B, conflict-free b128)
// ---------------------------------------------------------------------------
__device__ __forceinline__ void stage_tiles(int kt, int nbuf, int wave, int lane,
                                            const bf16_t* kg, const bf16_t* vg,
                                            bf16_t (*sK)[32][264],
                                            bf16_t (*sV)[256][40]) {
    // K tile: 16KB contiguous at byte offset kt*16384
#pragma unroll
    for (int j = 0; j < 8; ++j) {
        int i = wave * 32 + lane + j * 128;          // 16B chunk index 0..1023
        unsigned goff = (unsigned)(kt * 16384 + i * 16);
        unsigned lds  = (unsigned)(size_t)(&sK[nbuf][i >> 5][(i & 31) * 8]);
        asm volatile("global_load_async_to_lds_b128 %0, %1, %2"
                     :: "v"(lds), "v"(goff), "s"(kg) : "memory");
    }
    // V^T tile: 256 rows x 64B, global row stride 8192B, col offset kt*64
#pragma unroll
    for (int j = 0; j < 8; ++j) {
        int i = wave * 32 + lane + j * 128;
        int d = i >> 2;
        unsigned goff = (unsigned)(d * 8192 + kt * 64 + (i & 3) * 16);
        unsigned lds  = (unsigned)(size_t)(&sV[nbuf][d][(i & 3) * 8]);
        asm volatile("global_load_async_to_lds_b128 %0, %1, %2"
                     :: "v"(lds), "v"(goff), "s"(vg) : "memory");
    }
}

__global__ void __launch_bounds__(128, 1)
flash_attn(const bf16_t* __restrict__ qb, const bf16_t* __restrict__ kb,
           const bf16_t* __restrict__ vtb, bf16_t* __restrict__ attb) {
    const int tid = threadIdx.x;
    const int wave = tid >> 5, lane = tid & 31;
    const int half = lane >> 4, ln = lane & 15;
    const int bh = blockIdx.x;
    const int b = bh >> 2, h = bh & 3;
    const int qbase = blockIdx.y * 64 + wave * 16;

    __shared__ bf16_t sK[2][32][264];
    __shared__ bf16_t sV[2][256][40];
    __shared__ bf16_t pl[4][16][40];

    const size_t base = (size_t)bh * NN * DD;
    const bf16_t* kg = kb + base;                    // K tile base for this head
    const bf16_t* vg = vtb + (size_t)bh * DD * NN;   // V^T base for this head

    // Q fragments (A layout) for all 8 K-chunks, kept in registers
    Frag qf[8];
    {
        const bf16_t* qrow = qb + base + (size_t)(qbase + ln) * DD;
#pragma unroll
        for (int kc = 0; kc < 8; ++kc) {
            qf[kc].u[0] = *(const v4u*)(qrow + kc * 32 + half * 16);
            qf[kc].u[1] = *(const v4u*)(qrow + kc * 32 + half * 16 + 8);
        }
    }

    v8f acc[16];
#pragma unroll
    for (int nt = 0; nt < 16; ++nt)
#pragma unroll
        for (int r = 0; r < 8; ++r) acc[nt][r] = 0.f;

    float mrow[8], lrow[8];
#pragma unroll
    for (int r = 0; r < 8; ++r) { mrow[r] = -1e30f; lrow[r] = 0.f; }

    stage_tiles(0, 0, wave, lane, kg, vg, sK, sV);

    for (int kt = 0; kt < NN / 32; ++kt) {
        const int buf = kt & 1;
        if (kt + 1 < NN / 32) {
            stage_tiles(kt + 1, buf ^ 1, wave, lane, kg, vg, sK, sV);
            asm volatile("s_wait_asynccnt 0x10" ::: "memory");   // current buf done
        } else {
            asm volatile("s_wait_asynccnt 0x0" ::: "memory");
        }
        __syncthreads();

        // ---- scores: two 16x16 subtiles over 32 keys (B from LDS) ----
        v8f S[2];
#pragma unroll
        for (int sub = 0; sub < 2; ++sub) {
            v8f s;
#pragma unroll
            for (int r = 0; r < 8; ++r) s[r] = 0.f;
            const bf16_t* krow = &sK[buf][sub * 16 + ln][0];
#pragma unroll
            for (int kc = 0; kc < 8; ++kc) {
                Frag bf;
                bf.u[0] = *(const v4u*)(krow + kc * 32 + half * 16);
                bf.u[1] = *(const v4u*)(krow + kc * 32 + half * 16 + 8);
                s = wmma_bf16(qf[kc].v, bf.v, s);
            }
            S[sub] = s;
        }

        // ---- online softmax update (row stats within 16-lane halves) ----
        float sc[8];
#pragma unroll
        for (int r = 0; r < 8; ++r) {
            float t = fmaxf(S[0][r], S[1][r]);
            t = fmaxf(t, __shfl_xor(t, 1, 32));
            t = fmaxf(t, __shfl_xor(t, 2, 32));
            t = fmaxf(t, __shfl_xor(t, 4, 32));
            t = fmaxf(t, __shfl_xor(t, 8, 32));
            float mnew = fmaxf(mrow[r], t);
            sc[r] = __builtin_amdgcn_exp2f((mrow[r] - mnew) * LOG2E);
            float p0 = __builtin_amdgcn_exp2f((S[0][r] - mnew) * LOG2E);
            float p1 = __builtin_amdgcn_exp2f((S[1][r] - mnew) * LOG2E);
            S[0][r] = p0;
            S[1][r] = p1;
            float rs = p0 + p1;
            rs += __shfl_xor(rs, 1, 32);
            rs += __shfl_xor(rs, 2, 32);
            rs += __shfl_xor(rs, 4, 32);
            rs += __shfl_xor(rs, 8, 32);
            lrow[r] = lrow[r] * sc[r] + rs;
            mrow[r] = mnew;
        }
#pragma unroll
        for (int nt = 0; nt < 16; ++nt)
#pragma unroll
            for (int r = 0; r < 8; ++r) acc[nt][r] *= sc[r];

        // ---- reformat P (C/D layout) -> A layout via LDS ----
#pragma unroll
        for (int r = 0; r < 8; ++r) {
            pl[wave][8 * half + r][ln]      = f2bf(S[0][r]);
            pl[wave][8 * half + r][16 + ln] = f2bf(S[1][r]);
        }
        Frag pa;
        pa.u[0] = *(const v4u*)&pl[wave][ln][half * 16];
        pa.u[1] = *(const v4u*)&pl[wave][ln][half * 16 + 8];

        // ---- PV: acc += P (16x32) @ V (32x256), B tiles from LDS ----
#pragma unroll
        for (int nt = 0; nt < 16; ++nt) {
            const bf16_t* vp = &sV[buf][nt * 16 + ln][half * 16];
            Frag bv;
            bv.u[0] = *(const v4u*)(vp);
            bv.u[1] = *(const v4u*)(vp + 8);
            acc[nt] = wmma_bf16(pa.v, bv.v, acc[nt]);
        }
        __syncthreads();   // protect buf before it is restaged next iteration
    }

    // ---- normalize and store concat-head layout [b][n][h*256 + d] ----
    float rl[8];
#pragma unroll
    for (int r = 0; r < 8; ++r) rl[r] = __builtin_amdgcn_rcpf(lrow[r]);
#pragma unroll
    for (int nt = 0; nt < 16; ++nt)
#pragma unroll
        for (int r = 0; r < 8; ++r) {
            int gm = qbase + 8 * half + r;
            attb[((size_t)b * NN + gm) * KO + h * DD + nt * 16 + ln] =
                f2bf(acc[nt][r] * rl[r]);
        }
}

// ---------------------------------------------------------------------------
// Output projection: emb = att (16384x1024) @ Wo (1024x256) + bo + point_emb
// grid = BN/64, block = 128. Weight staged to LDS in four 128KB K-slabs.
// ---------------------------------------------------------------------------
__global__ void __launch_bounds__(128, 1)
out_proj(const bf16_t* __restrict__ A, const bf16_t* __restrict__ WoT,
         const float* __restrict__ bo, const float* __restrict__ pe,
         float* __restrict__ emb) {
    const int tid = threadIdx.x;
    const int wave = tid >> 5, lane = tid & 31;
    const int half = lane >> 4, ln = lane & 15;
    const int rowbase = blockIdx.x * 64 + wave * 16;

    __shared__ bf16_t sW[256][264];

    const bf16_t* arow = A + (size_t)(rowbase + ln) * KO;

    v8f acc[16];
#pragma unroll
    for (int nt = 0; nt < 16; ++nt)
#pragma unroll
        for (int r = 0; r < 8; ++r) acc[nt][r] = 0.f;

    for (int s = 0; s < 4; ++s) {
        __syncthreads();                 // previous slab fully consumed
        stage_w_panel(WoT, tid, sW, KO * 2, s * 512);

        Frag af[8];
#pragma unroll
        for (int kc = 0; kc < 8; ++kc) {
            const bf16_t* ap = arow + (s * 8 + kc) * 32;
            af[kc].u[0] = *(const v4u*)(ap + half * 16);
            af[kc].u[1] = *(const v4u*)(ap + half * 16 + 8);
        }

        asm volatile("s_wait_asynccnt 0x0" ::: "memory");
        __syncthreads();

#pragma unroll
        for (int kc = 0; kc < 8; ++kc) {
#pragma unroll
            for (int nt = 0; nt < 16; ++nt) {
                const bf16_t* wp = &sW[nt * 16 + ln][kc * 32 + half * 16];
                Frag bf;
                bf.u[0] = *(const v4u*)(wp);
                bf.u[1] = *(const v4u*)(wp + 8);
                acc[nt] = wmma_bf16(af[kc].v, bf.v, acc[nt]);
            }
        }
    }
#pragma unroll
    for (int nt = 0; nt < 16; ++nt) {
        int col = nt * 16 + ln;
        float bv = bo[col];
#pragma unroll
        for (int r = 0; r < 8; ++r) {
            int gm = rowbase + 8 * half + r;
            emb[(size_t)gm * DD + col] = acc[nt][r] + bv + pe[(size_t)gm * DD + col];
        }
    }
}

// ---------------------------------------------------------------------------
// FFN head: sigmoid(LN(relu(LN(relu(emb@W1+b1))@W2+b2))@W3+b3)
// one block (128 threads) per token
// ---------------------------------------------------------------------------
__device__ inline float block_sum128(float v, float* scratch, int tid) {
    v += __shfl_xor(v, 1, 32);
    v += __shfl_xor(v, 2, 32);
    v += __shfl_xor(v, 4, 32);
    v += __shfl_xor(v, 8, 32);
    v += __shfl_xor(v, 16, 32);
    if ((tid & 31) == 0) scratch[tid >> 5] = v;
    __syncthreads();
    float s = scratch[0] + scratch[1] + scratch[2] + scratch[3];
    __syncthreads();
    return s;
}

__global__ __launch_bounds__(128) void ffn_head(const float* __restrict__ emb,
                                                const float* __restrict__ W1,
                                                const float* __restrict__ b1,
                                                const float* __restrict__ g1,
                                                const float* __restrict__ be1,
                                                const float* __restrict__ W2,
                                                const float* __restrict__ b2,
                                                const float* __restrict__ g2,
                                                const float* __restrict__ be2,
                                                const float* __restrict__ W3,
                                                const float* __restrict__ b3,
                                                float* __restrict__ act) {
    __shared__ float se[256];
    __shared__ float sh1[128];
    __shared__ float sh2[64];
    __shared__ float scratch[4];
    const int row = blockIdx.x;
    const int t = threadIdx.x;

    se[t]       = emb[(size_t)row * 256 + t];
    se[t + 128] = emb[(size_t)row * 256 + 128 + t];
    __syncthreads();

    // layer 1: 256 -> 128, relu, LN
    float h = b1[t];
    for (int d = 0; d < 256; ++d) h += se[d] * W1[d * 128 + t];
    h = fmaxf(h, 0.f);
    float mu  = block_sum128(h, scratch, t) * (1.f / 128.f);
    float m2  = block_sum128(h * h, scratch, t) * (1.f / 128.f);
    float var = m2 - mu * mu;
    sh1[t] = (h - mu) * __builtin_amdgcn_rsqf(var + EPS) * g1[t] + be1[t];
    __syncthreads();

    // layer 2: 128 -> 64, relu, LN
    float h2 = 0.f;
    if (t < 64) {
        h2 = b2[t];
        for (int d = 0; d < 128; ++d) h2 += sh1[d] * W2[d * 64 + t];
        h2 = fmaxf(h2, 0.f);
    }
    float mu2  = block_sum128(t < 64 ? h2 : 0.f, scratch, t) * (1.f / 64.f);
    float m22  = block_sum128(t < 64 ? h2 * h2 : 0.f, scratch, t) * (1.f / 64.f);
    float var2 = m22 - mu2 * mu2;
    if (t < 64)
        sh2[t] = (h2 - mu2) * __builtin_amdgcn_rsqf(var2 + EPS) * g2[t] + be2[t];
    __syncthreads();

    // layer 3: 64 -> 1, sigmoid
    float c = (t < 64) ? sh2[t] * W3[t] : 0.f;
    float tot = block_sum128(c, scratch, t);
    if (t == 0) {
        float x = tot + b3[0];
        act[row] = 1.f / (1.f + __builtin_amdgcn_exp2f(-x * LOG2E));
    }
}

// ---------------------------------------------------------------------------
// launcher
// ---------------------------------------------------------------------------
extern "C" void kernel_launch(void* const* d_in, const int* in_sizes, int n_in,
                              void* d_out, int out_size, void* d_ws, size_t ws_size,
                              hipStream_t stream) {
    const float* img_emb   = (const float*)d_in[0];
    const float* point_emb = (const float*)d_in[1];
    const float* Wq = (const float*)d_in[2];
    const float* bq = (const float*)d_in[3];
    const float* Wk = (const float*)d_in[4];
    const float* bk = (const float*)d_in[5];
    const float* Wv = (const float*)d_in[6];
    const float* bv = (const float*)d_in[7];
    const float* Wo = (const float*)d_in[8];
    const float* bo = (const float*)d_in[9];
    const float* pos = (const float*)d_in[10];
    const float* W1 = (const float*)d_in[11];
    const float* b1 = (const float*)d_in[12];
    const float* g1 = (const float*)d_in[13];
    const float* be1 = (const float*)d_in[14];
    const float* W2 = (const float*)d_in[15];
    const float* b2 = (const float*)d_in[16];
    const float* g2 = (const float*)d_in[17];
    const float* be2 = (const float*)d_in[18];
    const float* W3 = (const float*)d_in[19];
    const float* b3 = (const float*)d_in[20];

    float* out_act = (float*)d_out;                 // [B*N]
    float* out_emb = out_act + (size_t)BN;          // [B*N, 256]

    // workspace carve
    char* w = (char*)d_ws;
    bf16_t* x_pt = (bf16_t*)w;      w += (size_t)BN * DD * 2;
    bf16_t* x_img = (bf16_t*)w;     w += (size_t)BN * DD * 2;
    bf16_t* x_imgpos = (bf16_t*)w;  w += (size_t)BN * DD * 2;
    bf16_t* WqT = (bf16_t*)w;       w += (size_t)HH * DD * DD * 2;
    bf16_t* WkT = (bf16_t*)w;       w += (size_t)HH * DD * DD * 2;
    bf16_t* WvT = (bf16_t*)w;       w += (size_t)HH * DD * DD * 2;
    bf16_t* WoT = (bf16_t*)w;       w += (size_t)DD * KO * 2;
    bf16_t* qb  = (bf16_t*)w;       w += (size_t)BB * HH * NN * DD * 2;
    bf16_t* kb  = (bf16_t*)w;       w += (size_t)BB * HH * NN * DD * 2;
    bf16_t* vtb = (bf16_t*)w;       w += (size_t)BB * HH * NN * DD * 2;
    bf16_t* attb = (bf16_t*)w;      w += (size_t)BN * KO * 2;

    // 1) inputs to bf16 (+ img transpose, + pos add)
    cvt_bf16<<<2048, 256, 0, stream>>>(point_emb, x_pt, (size_t)BN * DD);
    transpose_img<<<dim3(NN / 32, DD / 32, BB), dim3(32, 8), 0, stream>>>(
        img_emb, pos, x_img, x_imgpos);

    // 2) weights to bf16 transposed [out][in]
    wtrans<<<1024, 256, 0, stream>>>(Wq, WqT, DD, DD, HH);
    wtrans<<<1024, 256, 0, stream>>>(Wk, WkT, DD, DD, HH);
    wtrans<<<1024, 256, 0, stream>>>(Wv, WvT, DD, DD, HH);
    wtrans<<<1024, 256, 0, stream>>>(Wo, WoT, KO, DD, 1);

    // 3) QKV projections (q scaled by 1/sqrt(D); v stored transposed [d][n])
    proj_gemm<<<dim3(BB * HH, NN / 64), 128, 0, stream>>>(x_pt, WqT, bq, qb,
                                                          0.0625f, 0);
    proj_gemm<<<dim3(BB * HH, NN / 64), 128, 0, stream>>>(x_img, WkT, bk, kb,
                                                          1.0f, 0);
    proj_gemm<<<dim3(BB * HH, NN / 64), 128, 0, stream>>>(x_imgpos, WvT, bv, vtb,
                                                          1.0f, 1);

    // 4) flash attention
    flash_attn<<<dim3(BB * HH, NN / 64), 128, 0, stream>>>(qb, kb, vtb, attb);

    // 5) output projection + residual (fp32 emb output)
    out_proj<<<BN / 64, 128, 0, stream>>>(attb, WoT, bo, point_emb, out_emb);

    // 6) FFN head -> act
    ffn_head<<<BN, 128, 0, stream>>>(out_emb, W1, b1, g1, be1, W2, b2, g2, be2,
                                     W3, b3, out_act);
}
